// TimeceptionBlockV2_16458314679014
// MI455X (gfx1250) — matrix operand order repaired and checked by reading, not compiled, not run on hardware
//
#include <hip/hip_runtime.h>

// ---------------------------------------------------------------------------
// Timeception block for MI455X (gfx1250), wave32 + v_wmma_f32_16x16x32_bf16
// ---------------------------------------------------------------------------

typedef __attribute__((ext_vector_type(16))) __bf16 v16bf;
typedef __attribute__((ext_vector_type(8)))  float  v8f;

#define C_DIM  1024
#define IC_DIM 512
#define HW     196
#define NPAD   224   // 7 tiles of 32
#define BT_DIM 128
#define T_DIM  8

static __device__ __forceinline__ unsigned short f2bf(float f) {
  unsigned int u = __float_as_uint(f);
  unsigned int r = 0x7FFFu + ((u >> 16) & 1u);
  return (unsigned short)((u + r) >> 16);
}

// ---------------- f32 -> bf16 weight convert ----------------
__global__ void cvt_f32_bf16(const float* __restrict__ in,
                             unsigned short* __restrict__ out, int n) {
  int i = blockIdx.x * blockDim.x + threadIdx.x;
  if (i < n) out[i] = f2bf(in[i]);
}

// ---------------- fold 4 depthwise temporal convs into one 7-tap ----------------
__global__ void build_weff(const float* __restrict__ wb1, const float* __restrict__ bb1,
                           const float* __restrict__ wb2, const float* __restrict__ bb2,
                           const float* __restrict__ wb3, const float* __restrict__ bb3,
                           const float* __restrict__ wb4, const float* __restrict__ bb4,
                           float* __restrict__ weff, float* __restrict__ beff) {
  int ic = blockIdx.x * blockDim.x + threadIdx.x;
  if (ic >= IC_DIM) return;
#pragma unroll
  for (int j = 0; j < 7; ++j) {
    int d = j - 3;
    float w = wb4[ic * 7 + j];
    if (d >= -2 && d <= 2) w += wb3[ic * 5 + (d + 2)];
    if (d >= -1 && d <= 1) w += wb2[ic * 3 + (d + 1)];
    if (d == 0)            w += wb1[ic];
    weff[ic * 8 + j] = w;
  }
  weff[ic * 8 + 7] = 0.f;
  beff[ic] = bb1[ic] + bb2[ic] + bb3[ic] + bb4[ic];
}

// ---------------- x[img][c][hw] (f32) -> xT[img][p<224][c] (bf16) ----------------
__global__ void transpose_x(const float* __restrict__ x, unsigned short* __restrict__ xT) {
  __shared__ float lds[32][33];
  int bid = blockIdx.x;                 // 128 * 32 * 7 blocks
  int img = bid / (32 * 7);
  int r   = bid % (32 * 7);
  int ct = r / 7, ht = r % 7;
  int tx = threadIdx.x & 31, ty = threadIdx.x >> 5;
  const float* xs = x + (size_t)img * C_DIM * HW;
#pragma unroll
  for (int j = 0; j < 4; ++j) {
    int c  = ct * 32 + ty + j * 8;
    int hw = ht * 32 + tx;
    lds[ty + j * 8][tx] = (hw < HW) ? xs[(size_t)c * HW + hw] : 0.f;
  }
  __syncthreads();
  unsigned short* xo = xT + (size_t)img * NPAD * C_DIM;
#pragma unroll
  for (int j = 0; j < 4; ++j) {
    int pl = ty + j * 8;
    int p  = ht * 32 + pl;
    int c  = ct * 32 + tx;
    xo[(size_t)p * C_DIM + c] = f2bf(lds[tx][pl]);
  }
}

// ---------------- 7-tap temporal conv + transpose: y -> tmixT bf16 ----------------
// y:  [n=(b*8+t)][ic(512)][hw(196)] f32 ;  tT: [n][p<224][ic] bf16
__global__ void temporal_conv(const float* __restrict__ y,
                              const float* __restrict__ weff,
                              const float* __restrict__ beff,
                              unsigned short* __restrict__ tT) {
  __shared__ float lds[32][33];
  int bid = blockIdx.x;                 // 128 * 16 * 7 blocks
  int n   = bid / (16 * 7);
  int r   = bid % (16 * 7);
  int ct = r / 7, ht = r % 7;
  int b = n >> 3, t = n & 7;
  int tx = threadIdx.x & 31, ty = threadIdx.x >> 5;
#pragma unroll
  for (int j = 0; j < 4; ++j) {
    int ic = ct * 32 + ty + j * 8;
    int hw = ht * 32 + tx;
    float acc = 0.f;
    if (hw < HW) {
      acc = beff[ic];
#pragma unroll
      for (int d = 0; d < 7; ++d) {     // tap branch is uniform per block
        int tp = t + d - 3;
        if (tp >= 0 && tp < T_DIM)
          acc += weff[ic * 8 + d] *
                 y[((size_t)(b * T_DIM + tp) * IC_DIM + ic) * HW + hw];
      }
    }
    lds[ty + j * 8][tx] = acc;
  }
  __syncthreads();
  unsigned short* to = tT + (size_t)n * NPAD * IC_DIM;
#pragma unroll
  for (int j = 0; j < 4; ++j) {
    int pl = ty + j * 8;
    int p  = ht * 32 + pl;
    int ic = ct * 32 + tx;
    to[(size_t)p * IC_DIM + ic] = f2bf(lds[tx][pl]);
  }
}

// ---------------- WMMA GEMM: out[img][M][196] = A[M][K] @ Bt[img][n][K]^T (+bias,+resid) ----
template <int M, bool RES>
__device__ __forceinline__ void store_tile(const v8f& acc, int img, int mbase, int n,
                                           int lhi, const float* __restrict__ bias,
                                           const float* __restrict__ resid,
                                           float* __restrict__ out) {
  if (n >= HW) return;
#pragma unroll
  for (int v = 0; v < 8; ++v) {        // D layout: m = v + 8*lhi, n = lane%16
    int m = mbase + v + 8 * lhi;
    size_t idx = ((size_t)img * M + m) * (size_t)HW + n;
    float val = acc[v] + bias[m];
    if constexpr (RES) val += resid[idx];
    out[idx] = val;
  }
}

template <int M, int K, bool RES>
__global__ void gemm_wmma(const __bf16* __restrict__ A,   // [M][K] row-major bf16
                          const __bf16* __restrict__ Bt,  // [128][NPAD][K] bf16 (k contiguous)
                          const float*  __restrict__ bias,
                          const float*  __restrict__ resid,
                          float* __restrict__ out) {
  constexpr int MT = M / 32;
  const int lane = threadIdx.x & 31;
  const int wave = threadIdx.x >> 5;
  const int lrow = lane & 15;
  const int lhi  = lane >> 4;

  int gw = blockIdx.x * 8 + wave;               // 256 threads = 8 waves
  const int img  = gw / (MT * 7);
  int r          = gw % (MT * 7);
  const int mpos = r / 7, npos = r % 7;
  const int m0 = mpos * 32, n0 = npos * 32;

  // A fragment (16-bit A 16x32): lane m = lrow; k = {8*lhi..+7, 16+8*lhi..+7}
  const __bf16* Arow = A + (size_t)(m0 + lrow) * K;
  // B fragment (16-bit B 32x16): lane n = lrow; k = {8*lhi..+7, 16+8*lhi..+7}
  const __bf16* Brow = Bt + ((size_t)img * NPAD + n0 + lrow) * K;
  const int aoff = 8 * lhi;
  const int boff = 8 * lhi;

  v8f acc00 = {}, acc01 = {}, acc10 = {}, acc11 = {};
  union V { v16bf bf; uint4 u[2]; };

  for (int k0 = 0; k0 < K; k0 += 32) {
    V a0, a1, b0, b1;
    a0.u[0] = *(const uint4*)(Arow + k0 + aoff);
    a0.u[1] = *(const uint4*)(Arow + k0 + 16 + aoff);
    a1.u[0] = *(const uint4*)(Arow + (size_t)16 * K + k0 + aoff);
    a1.u[1] = *(const uint4*)(Arow + (size_t)16 * K + k0 + 16 + aoff);
    b0.u[0] = *(const uint4*)(Brow + k0 + boff);
    b0.u[1] = *(const uint4*)(Brow + k0 + boff + 16);
    b1.u[0] = *(const uint4*)(Brow + (size_t)16 * K + k0 + boff);
    b1.u[1] = *(const uint4*)(Brow + (size_t)16 * K + k0 + boff + 16);
    if (k0 + 32 < K) {                          // global_prefetch_b8 of next K-step
      __builtin_prefetch(Arow + k0 + 32, 0, 3);
      __builtin_prefetch(Brow + k0 + 32, 0, 3);
    }
    acc00 = __builtin_amdgcn_wmma_f32_16x16x32_bf16(false, a0.bf, false, b0.bf, (short)0, acc00, false, false);
    acc01 = __builtin_amdgcn_wmma_f32_16x16x32_bf16(false, a0.bf, false, b1.bf, (short)0, acc01, false, false);
    acc10 = __builtin_amdgcn_wmma_f32_16x16x32_bf16(false, a1.bf, false, b0.bf, (short)0, acc10, false, false);
    acc11 = __builtin_amdgcn_wmma_f32_16x16x32_bf16(false, a1.bf, false, b1.bf, (short)0, acc11, false, false);
  }

  store_tile<M, RES>(acc00, img, m0,      n0 + lrow,      lhi, bias, resid, out);
  store_tile<M, RES>(acc01, img, m0,      n0 + 16 + lrow, lhi, bias, resid, out);
  store_tile<M, RES>(acc10, img, m0 + 16, n0 + lrow,      lhi, bias, resid, out);
  store_tile<M, RES>(acc11, img, m0 + 16, n0 + 16 + lrow, lhi, bias, resid, out);
}

// ---------------------------------------------------------------------------
extern "C" void kernel_launch(void* const* d_in, const int* in_sizes, int n_in,
                              void* d_out, int out_size, void* d_ws, size_t ws_size,
                              hipStream_t stream) {
  const float* x   = (const float*)d_in[0];
  const float* w1  = (const float*)d_in[1];
  const float* b1  = (const float*)d_in[2];
  const float* wb1 = (const float*)d_in[3];
  const float* bb1 = (const float*)d_in[4];
  const float* wb2 = (const float*)d_in[5];
  const float* bb2 = (const float*)d_in[6];
  const float* wb3 = (const float*)d_in[7];
  const float* bb3 = (const float*)d_in[8];
  const float* wb4 = (const float*)d_in[9];
  const float* bb4 = (const float*)d_in[10];
  const float* w2  = (const float*)d_in[11];
  const float* b2  = (const float*)d_in[12];
  float* out = (float*)d_out;

  // workspace carve-up (all L2-resident by design; ~142 MB total)
  char* ws = (char*)d_ws;
  size_t off = 0;
  auto carve = [&](size_t bytes) -> void* {
    void* p = ws + off;
    off += (bytes + 255) & ~(size_t)255;
    return p;
  };
  unsigned short* A1   = (unsigned short*)carve((size_t)IC_DIM * C_DIM * 2);   // w1 bf16
  unsigned short* A2   = (unsigned short*)carve((size_t)C_DIM * IC_DIM * 2);   // w2 bf16
  float*          weff = (float*)carve((size_t)IC_DIM * 8 * 4);
  float*          beff = (float*)carve((size_t)IC_DIM * 4);
  unsigned short* xT   = (unsigned short*)carve((size_t)BT_DIM * NPAD * C_DIM * 2);
  float*          y    = (float*)carve((size_t)BT_DIM * IC_DIM * HW * 4);
  unsigned short* tT   = (unsigned short*)carve((size_t)BT_DIM * NPAD * IC_DIM * 2);

  // 1) weight converts + folded temporal kernel
  cvt_f32_bf16<<<(IC_DIM * C_DIM + 255) / 256, 256, 0, stream>>>(w1, A1, IC_DIM * C_DIM);
  cvt_f32_bf16<<<(C_DIM * IC_DIM + 255) / 256, 256, 0, stream>>>(w2, A2, C_DIM * IC_DIM);
  build_weff<<<2, 256, 0, stream>>>(wb1, bb1, wb2, bb2, wb3, bb3, wb4, bb4, weff, beff);

  // 2) stage x transposed to B-operand layout (bf16)
  transpose_x<<<BT_DIM * 32 * 7, 256, 0, stream>>>(x, xT);

  // 3) GEMM1: y = W1 @ x + b1      (128 imgs * 16 m-waves * 7 n-waves = 14336 waves)
  gemm_wmma<IC_DIM, C_DIM, false><<<BT_DIM * 16 * 7 / 8, 256, 0, stream>>>(
      (const __bf16*)A1, (const __bf16*)xT, b1, nullptr, y);

  // 4) fused 7-tap temporal conv + transpose -> tmixT (bf16, B-operand layout)
  temporal_conv<<<BT_DIM * 16 * 7, 256, 0, stream>>>(y, weff, beff, tT);

  // 5) GEMM2 + bias + residual: out = x + W2 @ tmix + b2   (28672 waves)
  gemm_wmma<C_DIM, IC_DIM, true><<<BT_DIM * 32 * 7 / 8, 256, 0, stream>>>(
      (const __bf16*)A2, (const __bf16*)tT, b2, x, out);
}
